// FlowAllocPlaceEmbModel_49065706389672
// MI455X (gfx1250) — matrix operand order repaired and checked by reading, not compiled, not run on hardware
//
#include <hip/hip_runtime.h>
#include <hip/hip_bf16.h>

typedef __attribute__((ext_vector_type(16))) __bf16 v16bf;
typedef __attribute__((ext_vector_type(8)))  __bf16 v8bf;
typedef __attribute__((ext_vector_type(8)))  float  v8f;

#define LN_EPS 1e-5f
#define D_DIM 64

// ---------------------------------------------------------------------------
// Kernel 1: LayerNorm both tables -> bf16 workspace. One wave32 per row,
// 2 elements per lane, wave shuffle reductions (wave32 on gfx1250).
// ---------------------------------------------------------------------------
__global__ __launch_bounds__(256) void ln_kernel(
    const float* __restrict__ Eo, const float* __restrict__ Ed,
    const float* __restrict__ go, const float* __restrict__ bo,
    const float* __restrict__ gd, const float* __restrict__ bd,
    __hip_bfloat16* __restrict__ Co, __hip_bfloat16* __restrict__ Cd, int N)
{
    int wave_global = (blockIdx.x * blockDim.x + threadIdx.x) >> 5;
    int lane = threadIdx.x & 31;
    if (wave_global >= 2 * N) return;

    int table = (wave_global >= N) ? 1 : 0;
    int row   = table ? (wave_global - N) : wave_global;
    const float* src = table ? Ed : Eo;
    const float* g   = table ? gd : go;
    const float* b   = table ? bd : bo;
    __hip_bfloat16* dst = table ? Cd : Co;

    const float2 x = *(const float2*)(src + (size_t)row * D_DIM + lane * 2);

    float s = x.x + x.y;
    #pragma unroll
    for (int o = 16; o > 0; o >>= 1) s += __shfl_xor(s, o, 32);
    float mean = s * (1.0f / D_DIM);

    float dx0 = x.x - mean, dx1 = x.y - mean;
    float v = dx0 * dx0 + dx1 * dx1;
    #pragma unroll
    for (int o = 16; o > 0; o >>= 1) v += __shfl_xor(v, o, 32);
    float inv = rsqrtf(v * (1.0f / D_DIM) + LN_EPS);

    float y0 = dx0 * inv * g[lane * 2 + 0] + b[lane * 2 + 0];
    float y1 = dx1 * inv * g[lane * 2 + 1] + b[lane * 2 + 1];

    dst[(size_t)row * D_DIM + lane * 2 + 0] = __float2bfloat16(y0);
    dst[(size_t)row * D_DIM + lane * 2 + 1] = __float2bfloat16(y1);
}

// ---------------------------------------------------------------------------
// Kernel 2: gather normalized query rows (bf16) into compact [B,64] buffers.
// q_out = Co[ids], q_in = Cd[ids]
// ---------------------------------------------------------------------------
__global__ __launch_bounds__(256) void gather_kernel(
    const int* __restrict__ ids,
    const __hip_bfloat16* __restrict__ Co, const __hip_bfloat16* __restrict__ Cd,
    __hip_bfloat16* __restrict__ Qo, __hip_bfloat16* __restrict__ Qi, int B)
{
    int t = blockIdx.x * blockDim.x + threadIdx.x;
    int per = B * D_DIM;
    if (t >= 2 * per) return;
    int which = (t >= per) ? 1 : 0;
    int rem = which ? (t - per) : t;
    int row = rem >> 6, col = rem & 63;
    int id = ids[row];
    if (which == 0) Qo[rem] = Co[(size_t)id * D_DIM + col];
    else            Qi[rem] = Cd[(size_t)id * D_DIM + col];
}

// ---------------------------------------------------------------------------
// Kernel 3: logits via WMMA bf16. One wave -> one 16x16 tile of out[B,N],
// K=64 as two v_wmma_f32_16x16x32_bf16 steps. 10 waves/block cover 160 cols.
// blockIdx.z selects outflow (Qo x Cd^T) vs inflow (Qi x Co^T).
// ---------------------------------------------------------------------------
__global__ __launch_bounds__(320) void gemm_kernel(
    const __hip_bfloat16* __restrict__ Qo, const __hip_bfloat16* __restrict__ Qi,
    const __hip_bfloat16* __restrict__ Co, const __hip_bfloat16* __restrict__ Cd,
    float* __restrict__ out, int N, int B)
{
    int wave = threadIdx.x >> 5;
    int lane = threadIdx.x & 31;
    int which = blockIdx.z;

    const __hip_bfloat16* Q = which ? Qi : Qo;
    const __hip_bfloat16* C = which ? Co : Cd;

    int n0 = blockIdx.x * 160 + wave * 16;
    int m0 = blockIdx.y * 16;
    if (n0 >= N) return;

    int lm = lane & 15;
    bool hi = lane >= 16;

    int m = m0 + lm; if (m > B - 1) m = B - 1;
    int n = n0 + lm; if (n > N - 1) n = N - 1;

    v8f acc;
    #pragma unroll
    for (int i = 0; i < 8; ++i) acc[i] = 0.0f;

    #pragma unroll
    for (int kb = 0; kb < 2; ++kb) {
        int kbase = kb * 32;

        // A (16x32 bf16): lane m holds K[kbase+aoff .. +15] (16 contiguous K)
        int aoff = hi ? 16 : 0;
        const __hip_bfloat16* aptr = Q + (size_t)m * D_DIM + kbase + aoff;
        v8bf a_lo = *(const v8bf*)(aptr);
        v8bf a_hi = *(const v8bf*)(aptr + 8);
        v16bf a;
        #pragma unroll
        for (int i = 0; i < 8; ++i) { a[i] = a_lo[i]; a[i + 8] = a_hi[i]; }

        // B (32x16 bf16): lane holds 16 contiguous K-halves of C row n
        int boff = hi ? 16 : 0;
        const __hip_bfloat16* bptr = C + (size_t)n * D_DIM + kbase + boff;
        v16bf bmat = *(const v16bf*)(bptr);

        acc = __builtin_amdgcn_wmma_f32_16x16x32_bf16(
            false, a, false, bmat, (short)0, acc, false, false);
    }

    // D layout: VGPR r -> M=m0+r (lanes 0-15) or M=m0+r+8 (lanes 16-31), N=n0+lm
    float* outp = out + (size_t)which * (size_t)B * (size_t)N;
    int col = n0 + lm;
    if (col < N && m0 + 15 < B) {
        // fast path: whole 16x16 tile in-bounds; 8 streaming stores,
        // row pointer bumped by N per result VGPR (no per-store mul/cmp)
        float* p = outp + (size_t)(m0 + (hi ? 8 : 0)) * (size_t)N + col;
        #pragma unroll
        for (int r = 0; r < 8; ++r) {
            *p = acc[r];
            p += N;
        }
    } else {
        #pragma unroll
        for (int r = 0; r < 8; ++r) {
            int mm = m0 + (hi ? r + 8 : r);
            if (col < N && mm < B)
                outp[(size_t)mm * (size_t)N + col] = acc[r];
        }
    }
}

// ---------------------------------------------------------------------------
// Kernel 4: row softmax over N for all 2*B rows (one block per row).
// ---------------------------------------------------------------------------
__global__ __launch_bounds__(256) void softmax_kernel(float* __restrict__ out, int N)
{
    __shared__ float sred[256];
    int tid = threadIdx.x;
    float* row = out + (size_t)blockIdx.x * (size_t)N;

    float lmax = -3.402823466e38f;
    for (int i = tid; i < N; i += 256) lmax = fmaxf(lmax, row[i]);
    sred[tid] = lmax; __syncthreads();
    for (int s = 128; s > 0; s >>= 1) {
        if (tid < s) sred[tid] = fmaxf(sred[tid], sred[tid + s]);
        __syncthreads();
    }
    float rmax = sred[0]; __syncthreads();

    float lsum = 0.0f;
    for (int i = tid; i < N; i += 256) lsum += __expf(row[i] - rmax);
    sred[tid] = lsum; __syncthreads();
    for (int s = 128; s > 0; s >>= 1) {
        if (tid < s) sred[tid] += sred[tid + s];
        __syncthreads();
    }
    float inv = 1.0f / sred[0]; __syncthreads();

    for (int i = tid; i < N; i += 256) row[i] = __expf(row[i] - rmax) * inv;
}

// ---------------------------------------------------------------------------
extern "C" void kernel_launch(void* const* d_in, const int* in_sizes, int n_in,
                              void* d_out, int out_size, void* d_ws, size_t ws_size,
                              hipStream_t stream) {
    const int*   ids = (const int*)  d_in[0];
    const float* Eo  = (const float*)d_in[1];
    const float* Ed  = (const float*)d_in[2];
    const float* go  = (const float*)d_in[3];
    const float* bo  = (const float*)d_in[4];
    const float* gd  = (const float*)d_in[5];
    const float* bd  = (const float*)d_in[6];
    float* out = (float*)d_out;

    int B = in_sizes[0];
    int N = in_sizes[1] / D_DIM;

    // workspace layout (bf16): Co[N,64] | Cd[N,64] | Qo[B,64] | Qi[B,64]
    __hip_bfloat16* Co = (__hip_bfloat16*)d_ws;
    __hip_bfloat16* Cd = Co + (size_t)N * D_DIM;
    __hip_bfloat16* Qo = Cd + (size_t)N * D_DIM;
    __hip_bfloat16* Qi = Qo + (size_t)B * D_DIM;

    // 1) LayerNorm tables -> bf16 (one wave32 per row, 8 rows per 256-thr block)
    int ln_blocks = (2 * N + 7) / 8;
    ln_kernel<<<ln_blocks, 256, 0, stream>>>(Eo, Ed, go, bo, gd, bd, Co, Cd, N);

    // 2) gather query rows
    int g_elems = 2 * B * D_DIM;
    gather_kernel<<<(g_elems + 255) / 256, 256, 0, stream>>>(ids, Co, Cd, Qo, Qi, B);

    // 3) logits: grid (N/160, B/16, 2), 10 waves per block, 16x16 WMMA tiles
    dim3 ggrid((N + 159) / 160, (B + 15) / 16, 2);
    gemm_kernel<<<ggrid, 320, 0, stream>>>(Qo, Qi, Co, Cd, out, N, B);

    // 4) softmax over each of the 2*B rows
    softmax_kernel<<<2 * B, 256, 0, stream>>>(out, N);
}
